// DotProductAttention_11948599018129
// MI455X (gfx1250) — matrix operand order repaired and checked by reading, not compiled, not run on hardware
//
#include <hip/hip_runtime.h>

typedef __attribute__((ext_vector_type(16))) _Float16 v16h;
typedef __attribute__((ext_vector_type(8)))  _Float16 v8h;
typedef __attribute__((ext_vector_type(8)))  float    v8f;
typedef __fp16 fp16x2 __attribute__((ext_vector_type(2)));

#define SEQ   2048
#define DIM   128
#define KT    32               // keys per iteration
#define NT    (SEQ/KT)         // 64 iterations
#define QW    32               // queries per wave (two 16-wide groups)
#define WAVES 8
#define QB    (QW*WAVES)       // 256 queries per workgroup
#define KSTR  (DIM + 8)        // Kl row stride (halves)
#define VSTR  (KT + 8)         // Vt row stride (halves)

union H2U  { fp16x2 h; unsigned u; _Float16 f[2]; };
union V8U  { v8h  v; unsigned d[4]; };
union V16U { v16h v; unsigned dw[8]; };

// pack two f32 -> one dword of two f16 (v_cvt_pk_rtz_f16_f32)
__device__ __forceinline__ unsigned pack2(float a, float b) {
    H2U t; t.h = __builtin_amdgcn_cvt_pkrtz(a, b); return t.u;
}

// Issue global loads for one 32x128 f32 tile of K and V (held in registers).
__device__ __forceinline__ void load_tile(const float* __restrict__ Kb,
                                          const float* __restrict__ Vb,
                                          int kt, int row, int c0,
                                          float4* kr, float4* vr)
{
    const float4* kp = reinterpret_cast<const float4*>(Kb + (size_t)(kt + row)*DIM + c0);
    const float4* vp = reinterpret_cast<const float4*>(Vb + (size_t)(kt + row)*DIM + c0);
#pragma unroll
    for (int u = 0; u < 4; ++u) { kr[u] = kp[u]; vr[u] = vp[u]; }
}

// Convert staged registers to f16 and write K row-major / V transposed into LDS buffer.
__device__ __forceinline__ void stage_tile(_Float16 (*Klb)[KSTR],
                                           _Float16 (*Vtb)[VSTR],
                                           int row, int c0,
                                           const float4* kr, const float4* vr)
{
    V8U lo, hi;
    lo.d[0] = pack2(kr[0].x, kr[0].y);
    lo.d[1] = pack2(kr[0].z, kr[0].w);
    lo.d[2] = pack2(kr[1].x, kr[1].y);
    lo.d[3] = pack2(kr[1].z, kr[1].w);
    hi.d[0] = pack2(kr[2].x, kr[2].y);
    hi.d[1] = pack2(kr[2].z, kr[2].w);
    hi.d[2] = pack2(kr[3].x, kr[3].y);
    hi.d[3] = pack2(kr[3].z, kr[3].w);
    *reinterpret_cast<v8h*>(&Klb[row][c0])     = lo.v;
    *reinterpret_cast<v8h*>(&Klb[row][c0 + 8]) = hi.v;

#pragma unroll
    for (int u = 0; u < 4; ++u) {
        H2U p, q;
        p.u = pack2(vr[u].x, vr[u].y);
        q.u = pack2(vr[u].z, vr[u].w);
        Vtb[c0 + 4*u + 0][row] = p.f[0];
        Vtb[c0 + 4*u + 1][row] = p.f[1];
        Vtb[c0 + 4*u + 2][row] = q.f[0];
        Vtb[c0 + 4*u + 3][row] = q.f[1];
    }
}

__global__ __launch_bounds__(256)
void fa_fwd_f16wmma(const float* __restrict__ Q,
                    const float* __restrict__ K,
                    const float* __restrict__ V,
                    float* __restrict__ O)
{
    // Double-buffered: K tile row-major [key][d], V tile transposed [d][key]
    __shared__ __align__(16) _Float16 Kl[2][KT][KSTR];
    __shared__ __align__(16) _Float16 Vt[2][DIM][VSTR];

    const int bh   = blockIdx.y;
    const int qblk = blockIdx.x;
    const int tid  = threadIdx.x;
    const int wave = tid >> 5;
    const int lane = tid & 31;
    const int ln   = lane & 15;     // column (query) within tile
    const int g    = lane >> 4;     // half-wave group

    const int q0 = qblk * QB + wave * QW;        // first query of this wave
    const size_t baseQ[2] = { ((size_t)bh * SEQ + q0 + ln) * DIM,
                              ((size_t)bh * SEQ + q0 + 16 + ln) * DIM };
    const float* Kb = K + (size_t)bh * SEQ * DIM;
    const float* Vb = V + (size_t)bh * SEQ * DIM;

    // 1/sqrt(128) * log2(e): softmax done in exp2 domain (raw v_exp_f32)
    const float qscale = 0.088388347648318447f * 1.4426950408889634f;

    // Resident Q^T B-fragments for both query groups: qf[w][j] covers d in [32j,32j+32)
    V16U qf[2][4];
#pragma unroll
    for (int w = 0; w < 2; ++w) {
        const float* Qrow = Q + baseQ[w];
#pragma unroll
        for (int j = 0; j < 4; ++j) {
            const float4* p4 = reinterpret_cast<const float4*>(Qrow + j*32 + g*16);
            float4 x0 = p4[0], x1 = p4[1], x2 = p4[2], x3 = p4[3];
            qf[w][j].dw[0] = pack2(x0.x*qscale, x0.y*qscale);
            qf[w][j].dw[1] = pack2(x0.z*qscale, x0.w*qscale);
            qf[w][j].dw[2] = pack2(x1.x*qscale, x1.y*qscale);
            qf[w][j].dw[3] = pack2(x1.z*qscale, x1.w*qscale);
            qf[w][j].dw[4] = pack2(x2.x*qscale, x2.y*qscale);
            qf[w][j].dw[5] = pack2(x2.z*qscale, x2.w*qscale);
            qf[w][j].dw[6] = pack2(x3.x*qscale, x3.y*qscale);
            qf[w][j].dw[7] = pack2(x3.z*qscale, x3.w*qscale);
        }
    }

    v8f zero8 = {0.f,0.f,0.f,0.f,0.f,0.f,0.f,0.f};
    v8f acc[2][8];                    // acc[w][dt][r] = O[q_w][16*dt + 8*g + r]
#pragma unroll
    for (int w = 0; w < 2; ++w)
#pragma unroll
        for (int dt = 0; dt < 8; ++dt) acc[w][dt] = zero8;
    float mrun[2] = { -1e30f, -1e30f };
    float lrun[2] = { 0.f, 0.f };

    const int row = tid >> 3;         // 0..31 : key row this thread stages
    const int c0  = (tid & 7) * 16;   // d-column group

    // ---- prologue: stage tile 0 into buffer 0
    float4 kr[4], vr[4];
    load_tile(Kb, Vb, 0, row, c0, kr, vr);
    stage_tile(Kl[0], Vt[0], row, c0, kr, vr);

    for (int t = 0; t < NT; ++t) {
        __syncthreads();
        const int cur = t & 1;
        const int nxt = cur ^ 1;

        // issue next tile's global loads now; they complete during compute
        if (t + 1 < NT) load_tile(Kb, Vb, (t+1)*KT, row, c0, kr, vr);

        _Float16 (*Klc)[KSTR] = Kl[cur];
        _Float16 (*Vtc)[VSTR] = Vt[cur];

        // ---- GEMM1: S^T = K * Q^T ; each K fragment reused by both query groups
        v8f sc[2][2];
        sc[0][0]=zero8; sc[0][1]=zero8; sc[1][0]=zero8; sc[1][1]=zero8;
#pragma unroll
        for (int h = 0; h < 2; ++h) {
#pragma unroll
            for (int j = 0; j < 4; ++j) {
                v8h a0 = *reinterpret_cast<const v8h*>(&Klc[16*h + ln][32*j + 16*g]);
                v8h a1 = *reinterpret_cast<const v8h*>(&Klc[16*h + ln][32*j + 16*g + 8]);
                v16h a;
#pragma unroll
                for (int e = 0; e < 8; ++e) { a[e]=a0[e]; a[8+e]=a1[e]; }
                sc[0][h] = __builtin_amdgcn_wmma_f32_16x16x32_f16(false, a, false, qf[0][j].v,
                                                                  (short)0, sc[0][h], false, false);
                sc[1][h] = __builtin_amdgcn_wmma_f32_16x16x32_f16(false, a, false, qf[1][j].v,
                                                                  (short)0, sc[1][h], false, false);
            }
        }
        // sc[w][0][r] = score(key kt+8g+r, q_w); sc[w][1][r] = score(key kt+16+8g+r, q_w)

        // ---- online softmax per query group (per query == per lane column)
        V16U bp[2];
#pragma unroll
        for (int w = 0; w < 2; ++w) {
            float tmax = -1e30f;
#pragma unroll
            for (int r = 0; r < 8; ++r) tmax = fmaxf(tmax, fmaxf(sc[w][0][r], sc[w][1][r]));
            tmax = fmaxf(tmax, __shfl_xor(tmax, 16, 32));
            float mnew = fmaxf(mrun[w], tmax);

            // rescale only when some lane's max advanced (wave-uniform branch, EXEC stays full)
            if (__any(mnew > mrun[w])) {
                float corr = __builtin_amdgcn_exp2f(mrun[w] - mnew);
                lrun[w] *= corr;
#pragma unroll
                for (int dt = 0; dt < 8; ++dt)
#pragma unroll
                    for (int r = 0; r < 8; ++r) acc[w][dt][r] *= corr;
            }
            mrun[w] = mnew;

            float p0[8], p1[8];
            float tsum = 0.f;
#pragma unroll
            for (int r = 0; r < 8; ++r) {
                p0[r] = __builtin_amdgcn_exp2f(sc[w][0][r] - mnew);
                p1[r] = __builtin_amdgcn_exp2f(sc[w][1][r] - mnew);
                tsum += p0[r] + p1[r];
            }
            tsum += __shfl_xor(tsum, 16, 32);
            lrun[w] += tsum;

            // P^T B-fragment: pack to f16, then a single 4-dword cross-half exchange.
            // z = g ? h0d : h1d ; zx = shfl_xor(z,16):
            //   g=0 lane: zx = partner h0d (keys  8..15)
            //   g=1 lane: zx = partner h1d (keys 16..23)
            unsigned h0d[4], h1d[4];
#pragma unroll
            for (int i = 0; i < 4; ++i) {
                h0d[i] = pack2(p0[2*i], p0[2*i+1]);
                h1d[i] = pack2(p1[2*i], p1[2*i+1]);
            }
#pragma unroll
            for (int i = 0; i < 4; ++i) {
                unsigned z  = g ? h0d[i] : h1d[i];
                unsigned zx = (unsigned)__shfl_xor((int)z, 16, 32);
                bp[w].dw[i]     = g ? zx     : h0d[i];  // keys 16g + 0..7
                bp[w].dw[4 + i] = g ? h1d[i] : zx;      // keys 16g + 8..15
            }
        }

        // ---- GEMM2: O^T += V^T * P^T ; each V fragment reused by both query groups
#pragma unroll
        for (int dt = 0; dt < 8; ++dt) {
            v8h a0 = *reinterpret_cast<const v8h*>(&Vtc[dt*16 + ln][16*g]);
            v8h a1 = *reinterpret_cast<const v8h*>(&Vtc[dt*16 + ln][16*g + 8]);
            v16h a;
#pragma unroll
            for (int e = 0; e < 8; ++e) { a[e]=a0[e]; a[8+e]=a1[e]; }
            acc[0][dt] = __builtin_amdgcn_wmma_f32_16x16x32_f16(false, a, false, bp[0].v,
                                                                (short)0, acc[0][dt], false, false);
            acc[1][dt] = __builtin_amdgcn_wmma_f32_16x16x32_f16(false, a, false, bp[1].v,
                                                                (short)0, acc[1][dt], false, false);
        }

        // ---- convert staged registers and fill the other buffer for next iteration
        if (t + 1 < NT) stage_tile(Kl[nxt], Vt[nxt], row, c0, kr, vr);
    }

    // ---- epilogue: O[q][d] = acc / l ; lane writes 8 consecutive d per tile
#pragma unroll
    for (int w = 0; w < 2; ++w) {
        const float invl = 1.0f / lrun[w];
        float* Orow = O + baseQ[w];
#pragma unroll
        for (int dt = 0; dt < 8; ++dt) {
            float4 o0 = { acc[w][dt][0]*invl, acc[w][dt][1]*invl,
                          acc[w][dt][2]*invl, acc[w][dt][3]*invl };
            float4 o1 = { acc[w][dt][4]*invl, acc[w][dt][5]*invl,
                          acc[w][dt][6]*invl, acc[w][dt][7]*invl };
            *reinterpret_cast<float4*>(Orow + dt*16 + 8*g)     = o0;
            *reinterpret_cast<float4*>(Orow + dt*16 + 8*g + 4) = o1;
        }
    }
}

extern "C" void kernel_launch(void* const* d_in, const int* in_sizes, int n_in,
                              void* d_out, int out_size, void* d_ws, size_t ws_size,
                              hipStream_t stream) {
    (void)in_sizes; (void)n_in; (void)out_size; (void)d_ws; (void)ws_size;
    const float* q = (const float*)d_in[0];
    const float* k = (const float*)d_in[1];
    const float* v = (const float*)d_in[2];
    float* o = (float*)d_out;

    dim3 grid(SEQ / QB, 4 * 16);   // 8 query blocks x (B*H)=64
    fa_fwd_f16wmma<<<grid, 256, 0, stream>>>(q, k, v, o);
}
